// MoeLayer_78984448573478
// MI455X (gfx1250) — matrix operand order repaired and checked by reading, not compiled, not run on hardware
//
#include <hip/hip_runtime.h>

// ---------------------------------------------------------------------------
// MoE layer for gfx1250 (MI455X): routed (top-2) bf16 WMMA FFN, f32 accum.
// B=4 S=2048 D=1024 F=4096 E=8 TOPK=2
// Tokens are scattered into per-expert lists; each FFN block processes a
// 64-token tile of ONE expert -> per-block weight stream is 16.8 MB (one
// expert) instead of 134 MB (all experts): ~4.3 GB total L2 traffic and
// 275 GFLOP of bf16 WMMA instead of 1.1 TFLOP dense.
// ---------------------------------------------------------------------------

#define DD 1024
#define FF 4096
#define EE 8
#define NTOK 8192
#define FC 128          // F-chunk per iteration
#define MT 64           // tokens per block (4 WMMA M-tiles)
#define NTHREADS 512    // 16 wave32
#define XS_STRIDE 1032  // 1024 + 8 bf16 pad: 16B-aligned rows, bank-spread
#define HS_STRIDE 136   // 128 + 8 bf16 pad

typedef __attribute__((ext_vector_type(16))) __bf16 v16bf;
typedef __attribute__((ext_vector_type(8)))  __bf16 v8bf;
typedef __attribute__((ext_vector_type(8)))  float  v8f;

static __device__ __forceinline__ v16bf cat8(v8bf lo, v8bf hi) {
  return __builtin_shufflevector(lo, hi, 0, 1, 2, 3, 4, 5, 6, 7,
                                 8, 9, 10, 11, 12, 13, 14, 15);
}
static __device__ __forceinline__ v16bf ldv16(const __bf16* p) {
  return cat8(*(const v8bf*)p, *(const v8bf*)(p + 8));
}
static __device__ __forceinline__ v8f wmma_bf16(v16bf a, v16bf b, v8f c) {
  // (neg_a, A, neg_b, B, c_mod, C, reuse_a, reuse_b)
  return __builtin_amdgcn_wmma_f32_16x16x32_bf16(false, a, false, b,
                                                 (short)0, c, false, false);
}

// ---------------------------------------------------------------------------
// Zero the per-expert token counters (ws is poisoned, not re-poisoned).
// ---------------------------------------------------------------------------
__global__ void init_kernel(int* cnt) {
  if (threadIdx.x < EE) cnt[threadIdx.x] = 0;
}

// ---------------------------------------------------------------------------
// Zero the result region of d_out (FFN accumulates with atomics).
// ---------------------------------------------------------------------------
__global__ void __launch_bounds__(256) zero_kernel(float4* out4) {
  size_t i = (size_t)blockIdx.x * 256 + threadIdx.x;
  out4[i] = make_float4(0.f, 0.f, 0.f, 0.f);
}

// ---------------------------------------------------------------------------
// One-time f32 -> bf16 weight conversion with transpose:
//   w1t[e][f][d] = w1[e][d][f]   (B columns for GEMM1 become contiguous in d)
//   w2t[e][d][f] = w2[e][f][d]   (B columns for GEMM2 become contiguous in f)
// ---------------------------------------------------------------------------
__global__ void __launch_bounds__(256) convert_kernel(
    const float* __restrict__ w1, const float* __restrict__ w2,
    __bf16* __restrict__ w1t, __bf16* __restrict__ w2t) {
  size_t idx = (size_t)blockIdx.x * 256 + threadIdx.x;  // 0 .. E*D*F-1
  size_t e = idx / ((size_t)FF * DD);
  size_t r = idx % ((size_t)FF * DD);
  size_t f = r / DD, d = r % DD;
  w1t[idx] = (__bf16)w1[(e * DD + d) * FF + f];
  size_t d2 = r / FF, f2 = r % FF;
  w2t[idx] = (__bf16)w2[(e * FF + f2) * DD + d2];
}

// ---------------------------------------------------------------------------
// Gate: one wave32 per token. logits[8] -> full softmax (block-local partial
// sums for deterministic KL), top-2 -> renormalized softmax -> scatter
// (token, weight) into per-expert lists via atomic counters. The scatter
// order may vary, but each token's computation and the 2-way output sum are
// order-independent, so the final output is deterministic.
// ---------------------------------------------------------------------------
__global__ void __launch_bounds__(256) gate_kernel(
    const float* __restrict__ X, const float* __restrict__ gw,
    int* __restrict__ tokidx, float* __restrict__ cw, int* __restrict__ cnt,
    float* __restrict__ partial) {
  __shared__ float sg[8][EE];
  const int tid = threadIdx.x;
  const int wave = tid >> 5;
  const int lane = tid & 31;
  const int tok = blockIdx.x * 8 + wave;
  const float* x = X + (size_t)tok * DD;

  float p[EE];
#pragma unroll
  for (int q = 0; q < EE; ++q) p[q] = 0.0f;

  for (int d = lane; d < DD; d += 32) {
    float xv = x[d];
    const float* g = gw + (size_t)d * EE;
#pragma unroll
    for (int q = 0; q < EE; ++q) p[q] += xv * g[q];
  }
#pragma unroll
  for (int q = 0; q < EE; ++q) {
    for (int off = 16; off > 0; off >>= 1) p[q] += __shfl_xor(p[q], off, 32);
  }

  if (lane == 0) {
    float mx = p[0];
    int i1 = 0;
    for (int q = 1; q < EE; ++q)
      if (p[q] > mx) { mx = p[q]; i1 = q; }
    float pe[EE], se = 0.0f;
    for (int q = 0; q < EE; ++q) { pe[q] = __expf(p[q] - mx); se += pe[q]; }
    float inv = 1.0f / se;
    for (int q = 0; q < EE; ++q) sg[wave][q] = pe[q] * inv;

    float mx2 = -3.4e38f;
    int i2 = 0;
    for (int q = 0; q < EE; ++q)
      if (q != i1 && p[q] > mx2) { mx2 = p[q]; i2 = q; }
    float pa = 1.0f / (1.0f + __expf(mx2 - mx));  // softmax over top-2

    int pos1 = atomicAdd(&cnt[i1], 1);
    tokidx[i1 * NTOK + pos1] = tok;
    cw[i1 * NTOK + pos1] = pa;
    int pos2 = atomicAdd(&cnt[i2], 1);
    tokidx[i2 * NTOK + pos2] = tok;
    cw[i2 * NTOK + pos2] = 1.0f - pa;
  }
  __syncthreads();
  if (tid < EE) {
    float s = 0.0f;
    for (int w = 0; w < 8; ++w) s += sg[w][tid];
    partial[blockIdx.x * EE + tid] = s;
  }
}

// ---------------------------------------------------------------------------
// KL: deterministic fixed-order reduction of per-block softmax sums.
// kl = sum_e u*(log u - log mean_e) / E, u = 1/E.
// ---------------------------------------------------------------------------
__global__ void kl_kernel(const float* __restrict__ partial,
                          float* __restrict__ klout) {
  if (threadIdx.x == 0) {
    float kl = 0.0f;
    for (int q = 0; q < EE; ++q) {
      float s = 0.0f;
      for (int b = 0; b < NTOK / 8; ++b) s += partial[b * EE + q];
      float mean = s * (1.0f / (float)NTOK);
      kl += 0.125f * (__logf(0.125f) - __logf(mean));
    }
    klout[0] = kl * 0.125f;
  }
}

// ---------------------------------------------------------------------------
// Routed FFN: block (tile, expert) processes 64 tokens of expert e.
// combine weight folded into H rows; output accumulated with f32 atomics
// (exactly 2 contributions per element -> order-independent result).
// GEMM1: wave w -> (column tile w&7, M-pair w>>3), 2 hacc tiles.
// GEMM2: wave w owns output columns [w*64, w*64+64) for all 4 M-tiles.
// ---------------------------------------------------------------------------
__global__ void __launch_bounds__(NTHREADS) ffn_kernel(
    const float* __restrict__ X, const __bf16* __restrict__ w1t,
    const __bf16* __restrict__ w2t, const int* __restrict__ tokidx,
    const float* __restrict__ cw, const int* __restrict__ cnt,
    float* __restrict__ out) {
  const int e = blockIdx.y;
  const int n_e = cnt[e];
  const int base = blockIdx.x * MT;
  if (base >= n_e) return;  // uniform early exit (before any barrier)
  int nv = n_e - base;
  if (nv > MT) nv = MT;

  __shared__ __align__(16) __bf16 Xs[MT * XS_STRIDE];  // ~129 KB
  __shared__ __align__(16) __bf16 Hs[MT * HS_STRIDE];  // ~17 KB
  __shared__ int tki[MT];
  __shared__ float cws[MT];

  const int tid = threadIdx.x;
  const int wave = tid >> 5;    // 0..15
  const int lane = tid & 31;
  const int nloc = lane & 15;   // column within 16-wide tile
  const int hlf = lane >> 4;    // lane half (K-split per ISA layout)
  const int ct = wave & 7;      // GEMM1 column tile within F-chunk
  const int mp = wave >> 3;     // GEMM1 M-pair (rows 32*mp .. 32*mp+31)

  // Stage this tile's token ids and combine weights.
  if (tid < MT) {
    int ok = tid < nv;
    tki[tid] = tokidx[e * NTOK + base + (ok ? tid : 0)];
    cws[tid] = ok ? cw[e * NTOK + base + tid] : 0.0f;  // 0 => H row = 0
  }
  __syncthreads();

  // Gather X rows [64 x 1024] f32 -> bf16 into LDS (4 rows per wave).
#pragma unroll
  for (int r = 0; r < MT / 16; ++r) {
    int m = wave * (MT / 16) + r;
    const float4* src = (const float4*)(X + (size_t)tki[m] * DD);
    __bf16* dst = &Xs[m * XS_STRIDE];
    for (int d4 = lane; d4 < DD / 4; d4 += 32) {
      float4 v = src[d4];
      dst[d4 * 4 + 0] = (__bf16)v.x;
      dst[d4 * 4 + 1] = (__bf16)v.y;
      dst[d4 * 4 + 2] = (__bf16)v.z;
      dst[d4 * 4 + 3] = (__bf16)v.w;
    }
  }
  __syncthreads();

  v8f oacc[4][4] = {};  // 64 x 64 f32 output accumulator per wave

  // GEMM1 A rows (this wave's M-pair)
  const __bf16* xrow0 = &Xs[(32 * mp + nloc) * XS_STRIDE];
  const __bf16* xrow1 = &Xs[(32 * mp + 16 + nloc) * XS_STRIDE];
  // GEMM2 A rows (all 4 M-tiles)
  const __bf16* hrow[4];
#pragma unroll
  for (int mt = 0; mt < 4; ++mt) hrow[mt] = &Hs[(mt * 16 + nloc) * HS_STRIDE];

  const __bf16* w1e = w1t + (size_t)e * FF * DD;  // [F][D]
  const __bf16* w2e = w2t + (size_t)e * DD * FF;  // [D][F]

  // combine weights for this wave's GEMM1 rows (folded into H)
  float svg[2][8];
#pragma unroll
  for (int mt = 0; mt < 2; ++mt)
#pragma unroll
    for (int v = 0; v < 8; ++v)
      svg[mt][v] = cws[32 * mp + mt * 16 + v + 8 * hlf];

  for (int fc0 = 0; fc0 < FF; fc0 += FC) {
    // ---- GEMM1: h[32x16] = X[rows 32mp..+31] @ W1[:, ct's 16 cols] ----
    const __bf16* bcol1 = w1e + (size_t)(fc0 + ct * 16 + nloc) * DD + 16 * hlf;
    v8f hacc[2] = {};
#pragma unroll 4
    for (int k = 0; k < DD; k += 32) {
      __builtin_prefetch(bcol1 + k + 256, 0, 3);
      v16bf b = ldv16(bcol1 + k);
      v16bf a0 = cat8(*(const v8bf*)(xrow0 + k + 16 * hlf),
                      *(const v8bf*)(xrow0 + k + 8 + 16 * hlf));
      v16bf a1 = cat8(*(const v8bf*)(xrow1 + k + 16 * hlf),
                      *(const v8bf*)(xrow1 + k + 8 + 16 * hlf));
      hacc[0] = wmma_bf16(a0, b, hacc[0]);
      hacc[1] = wmma_bf16(a1, b, hacc[1]);
    }
    // silu, fold combine weight into H rows, write to LDS
#pragma unroll
    for (int mt = 0; mt < 2; ++mt)
#pragma unroll
      for (int v = 0; v < 8; ++v) {
        float h = hacc[mt][v];
        h = h / (1.0f + __expf(-h));
        Hs[(32 * mp + mt * 16 + v + 8 * hlf) * HS_STRIDE + ct * 16 + nloc] =
            (__bf16)(h * svg[mt][v]);
      }
    __syncthreads();

    // ---- GEMM2: out[64x64] += H[64x128] @ W2[chunk, wave's 64 cols] ----
#pragma unroll
    for (int k = 0; k < FC; k += 32) {
      v16bf a[4];
#pragma unroll
      for (int mt = 0; mt < 4; ++mt)
        a[mt] = cat8(*(const v8bf*)(hrow[mt] + k + 16 * hlf),
                     *(const v8bf*)(hrow[mt] + k + 8 + 16 * hlf));
#pragma unroll
      for (int t = 0; t < 4; ++t) {
        const __bf16* bcol2 = w2e +
            (size_t)(wave * 64 + t * 16 + nloc) * FF + fc0 + k + 16 * hlf;
        __builtin_prefetch(bcol2 + 32, 0, 3);
        v16bf b = ldv16(bcol2);
#pragma unroll
        for (int mt = 0; mt < 4; ++mt)
          oacc[mt][t] = wmma_bf16(a[mt], b, oacc[mt][t]);
      }
    }
    __syncthreads();
  }

  // Writeback: scatter-accumulate to the token's output row.
  // C/D layout -> row m = mt*16 + v + 8*hlf, col within tile.
#pragma unroll
  for (int t = 0; t < 4; ++t) {
    int col = wave * 64 + t * 16 + nloc;
#pragma unroll
    for (int mt = 0; mt < 4; ++mt)
#pragma unroll
      for (int v = 0; v < 8; ++v) {
        int row = mt * 16 + v + 8 * hlf;
        if (row < nv)
          atomicAdd(&out[(size_t)tki[row] * DD + col], oacc[mt][t][v]);
      }
  }
}

// ---------------------------------------------------------------------------
extern "C" void kernel_launch(void* const* d_in, const int* in_sizes, int n_in,
                              void* d_out, int out_size, void* d_ws,
                              size_t ws_size, hipStream_t stream) {
  const float* X  = (const float*)d_in[0];   // [B,S,D] f32
  const float* gw = (const float*)d_in[1];   // [D,E]   f32
  const float* w1 = (const float*)d_in[2];   // [E,D,F] f32
  const float* w2 = (const float*)d_in[3];   // [E,F,D] f32
  float* out = (float*)d_out;                // [B*S*D] + kl scalar

  // Workspace layout (~134.8 MB):
  //   [0, 64M)    w1t bf16 [E][F][D]
  //   [64M, 128M) w2t bf16 [E][D][F]
  //   then        tokidx int[E][N], cw f32[E][N], partial f32[N/8][E],
  //               cnt int[E]
  char* ws = (char*)d_ws;
  const size_t wbytes = (size_t)EE * DD * FF * sizeof(__bf16);
  __bf16* w1t = (__bf16*)ws;
  __bf16* w2t = (__bf16*)(ws + wbytes);
  size_t off = 2 * wbytes;
  int* tokidx = (int*)(ws + off);
  off += (size_t)EE * NTOK * sizeof(int);
  float* cw = (float*)(ws + off);
  off += (size_t)EE * NTOK * sizeof(float);
  float* partial = (float*)(ws + off);
  off += (size_t)(NTOK / 8) * EE * sizeof(float);
  int* cnt = (int*)(ws + off);

  init_kernel<<<1, 32, 0, stream>>>(cnt);
  convert_kernel<<<(unsigned)(((size_t)EE * DD * FF) / 256), 256, 0, stream>>>(
      w1, w2, w1t, w2t);
  gate_kernel<<<NTOK / 8, 256, 0, stream>>>(X, gw, tokidx, cw, cnt, partial);
  kl_kernel<<<1, 32, 0, stream>>>(partial, out + (size_t)NTOK * DD);
  zero_kernel<<<(NTOK * DD) / 1024, 256, 0, stream>>>((float4*)out);
  ffn_kernel<<<dim3(NTOK / MT, EE), NTHREADS, 0, stream>>>(
      X, w1t, w2t, tokidx, cw, cnt, out);
}
